// RingAttention_17927193493891
// MI455X (gfx1250) — matrix operand order repaired and checked by reading, not compiled, not run on hardware
//
#include <hip/hip_runtime.h>

typedef __attribute__((ext_vector_type(16))) _Float16 v16h;
typedef __attribute__((ext_vector_type(8)))  _Float16 v8h;
typedef __attribute__((ext_vector_type(2)))  _Float16 v2h;
typedef __attribute__((ext_vector_type(2)))  __fp16   v2fp;  // cvt_pkrtz result type
typedef __attribute__((ext_vector_type(8)))  float    v8f;

#define SEQ 2048
#define DH  128
#define BQ  128   // q rows per workgroup (8 waves x 16 rows)
#define BK  64    // kv rows per tile
#define NW  8     // waves per workgroup

__device__ __forceinline__ v8f wmma16(v16h a, v16h b, v8f c) {
  // D = A(16x32 f16) * B(32x16 f16) + C(16x16 f32)
  return __builtin_amdgcn_wmma_f32_16x16x32_f16(
      false, a, false, b, (short)0, c, false, false);
}

// Assemble a 16x32 f16 A-fragment lane slice from LDS:
// elements {0..7} at p, elements {8..15} at p+8 (p already includes the
// per-half 16-element offset).
__device__ __forceinline__ v16h ldsA(const _Float16* p) {
  v8h lo = *(const v8h*)(p);
  v8h hh = *(const v8h*)(p + 8);
  v16h r;
#pragma unroll
  for (int i = 0; i < 8; ++i) { r[i] = lo[i]; r[8 + i] = hh[i]; }
  return r;
}

__global__ __launch_bounds__(256)
void fa2_causal_wmma_kernel(const float* __restrict__ Q,
                            const float* __restrict__ K,
                            const float* __restrict__ V,
                            float* __restrict__ O) {
  __shared__ alignas(64) _Float16 sK[BK][DH];    // K tile, row-major f16  (16 KB)
  __shared__ alignas(64) _Float16 sVt[DH][BK];   // V tile, transposed f16 (16 KB)

  const int tid   = threadIdx.x;
  const int lane  = tid & 31;
  const int wave  = tid >> 5;
  const int nlo   = lane & 15;
  const int hi    = lane >> 4;
  const int khalf = hi * 16;   // B-fragment k-half select
  const int dsel  = hi * 8;    // C-tile row-half select

  const int q0  = blockIdx.x * BQ;
  const int wq0 = q0 + wave * 16;
  const int r   = wq0 + nlo;           // this lane's q row (transposed layout)
  const size_t base = (size_t)blockIdx.y * SEQ * DH;

  // ---- Q as B-fragments: contiguous loads, pre-scaled by (1/sqrt(D))*log2(e) ----
  const float qscale = 0.08838834764831845f * 1.4426950408889634f;
  const float* qrow = Q + base + (size_t)r * DH;
  v16h qb[4];
#pragma unroll
  for (int f = 0; f < 4; ++f) {
    const int d0 = f * 32 + khalf;
    float4 x0 = *(const float4*)(qrow + d0);
    float4 x1 = *(const float4*)(qrow + d0 + 4);
    float4 x2 = *(const float4*)(qrow + d0 + 8);
    float4 x3 = *(const float4*)(qrow + d0 + 12);
    v16h qv;
    qv[0]=(_Float16)(x0.x*qscale);  qv[1]=(_Float16)(x0.y*qscale);
    qv[2]=(_Float16)(x0.z*qscale);  qv[3]=(_Float16)(x0.w*qscale);
    qv[4]=(_Float16)(x1.x*qscale);  qv[5]=(_Float16)(x1.y*qscale);
    qv[6]=(_Float16)(x1.z*qscale);  qv[7]=(_Float16)(x1.w*qscale);
    qv[8]=(_Float16)(x2.x*qscale);  qv[9]=(_Float16)(x2.y*qscale);
    qv[10]=(_Float16)(x2.z*qscale); qv[11]=(_Float16)(x2.w*qscale);
    qv[12]=(_Float16)(x3.x*qscale); qv[13]=(_Float16)(x3.y*qscale);
    qv[14]=(_Float16)(x3.z*qscale); qv[15]=(_Float16)(x3.w*qscale);
    qb[f] = qv;
  }

  const v8f vzero = {0.f,0.f,0.f,0.f,0.f,0.f,0.f,0.f};
  v8f o[8];                    // O^T tiles: rows = d, cols = q(lane)
#pragma unroll
  for (int t = 0; t < 8; ++t) o[t] = vzero;
  float m_w = -1e30f, l_w = 0.f;   // per-lane (per-q-row) softmax stats

  const int kv_end = q0 + BQ;

  for (int kv = 0; kv < kv_end; kv += BK) {
    // ---- cooperative staging: K row-major f16, V transposed f16 ----
    {
      const int rr = (tid >> 3) * 2;     // 0,2,...,62
      const int cc = (tid & 7) * 16;     // 0..112
      const float* kp = K + base + (size_t)(kv + rr) * DH + cc;
      const float* vp = V + base + (size_t)(kv + rr) * DH + cc;
      float t0[16], t1[16];
      *(float4*)(t0 + 0)  = *(const float4*)(kp + 0);
      *(float4*)(t0 + 4)  = *(const float4*)(kp + 4);
      *(float4*)(t0 + 8)  = *(const float4*)(kp + 8);
      *(float4*)(t0 + 12) = *(const float4*)(kp + 12);
      *(float4*)(t1 + 0)  = *(const float4*)(kp + DH + 0);
      *(float4*)(t1 + 4)  = *(const float4*)(kp + DH + 4);
      *(float4*)(t1 + 8)  = *(const float4*)(kp + DH + 8);
      *(float4*)(t1 + 12) = *(const float4*)(kp + DH + 12);
#pragma unroll
      for (int i = 0; i < 16; ++i) sK[rr][cc + i]     = (_Float16)t0[i];
#pragma unroll
      for (int i = 0; i < 16; ++i) sK[rr + 1][cc + i] = (_Float16)t1[i];

      *(float4*)(t0 + 0)  = *(const float4*)(vp + 0);
      *(float4*)(t0 + 4)  = *(const float4*)(vp + 4);
      *(float4*)(t0 + 8)  = *(const float4*)(vp + 8);
      *(float4*)(t0 + 12) = *(const float4*)(vp + 12);
      *(float4*)(t1 + 0)  = *(const float4*)(vp + DH + 0);
      *(float4*)(t1 + 4)  = *(const float4*)(vp + DH + 4);
      *(float4*)(t1 + 8)  = *(const float4*)(vp + DH + 8);
      *(float4*)(t1 + 12) = *(const float4*)(vp + DH + 12);
#pragma unroll
      for (int i = 0; i < 16; ++i) {
        v2h p;
        p[0] = (_Float16)t0[i];
        p[1] = (_Float16)t1[i];
        *(v2h*)&sVt[cc + i][rr] = p;
      }
    }
    __syncthreads();

    if (kv + BK < kv_end) {
      __builtin_prefetch(K + base + (size_t)(kv + BK) * DH + tid * 32, 0, 1);
      __builtin_prefetch(V + base + (size_t)(kv + BK) * DH + tid * 32, 0, 1);
    }

    if (kv <= wq0 + 15) {
      // ---- S^T = K * Q^T : 4 C-tiles (rows = kv, cols = q across lanes) ----
      v8f s[4];
#pragma unroll
      for (int t = 0; t < 4; ++t) s[t] = vzero;
#pragma unroll
      for (int f = 0; f < 4; ++f) {
#pragma unroll
        for (int t = 0; t < 4; ++t) {
          v16h ka = ldsA(&sK[t * 16 + nlo][f * 32 + khalf]);
          s[t] = wmma16(ka, qb[f], s[t]);
        }
      }

      // ---- causal mask (diagonal tiles only): element kv index =
      //      kv + 16t + 8*hi + j ; mask iff > r  <=>  16t + j > thr ----
      if (kv + BK - 1 > wq0) {
        const int thr = r - kv - dsel;
#pragma unroll
        for (int t = 0; t < 4; ++t) {
#pragma unroll
          for (int j = 0; j < 8; ++j) {
            s[t][j] = (16 * t + j > thr) ? -1e30f : s[t][j];
          }
        }
      }

      // ---- online softmax: in-lane tree + ONE half-wave shuffle ----
      float m4[4];
#pragma unroll
      for (int t = 0; t < 4; ++t) {
        float a = fmaxf(fmaxf(s[t][0], s[t][1]), fmaxf(s[t][2], s[t][3]));
        float b = fmaxf(fmaxf(s[t][4], s[t][5]), fmaxf(s[t][6], s[t][7]));
        m4[t] = fmaxf(a, b);
      }
      float mx = fmaxf(fmaxf(m4[0], m4[1]), fmaxf(m4[2], m4[3]));
      mx = fmaxf(mx, __shfl_xor(mx, 16));
      const float mn = fmaxf(m_w, mx);
      const float al = __builtin_amdgcn_exp2f(m_w - mn);
      m_w = mn;

#pragma unroll
      for (int t = 0; t < 4; ++t) {
#pragma unroll
        for (int j = 0; j < 8; ++j) {
          s[t][j] = __builtin_amdgcn_exp2f(s[t][j] - mn);  // masked -> 0
        }
      }

      float r4[4];
#pragma unroll
      for (int t = 0; t < 4; ++t) {
        float a = (s[t][0] + s[t][1]) + (s[t][2] + s[t][3]);
        float b = (s[t][4] + s[t][5]) + (s[t][6] + s[t][7]);
        r4[t] = a + b;
      }
      float rs = (r4[0] + r4[1]) + (r4[2] + r4[3]);
      rs += __shfl_xor(rs, 16);
      l_w = l_w * al + rs;

      // ---- rescale O^T by per-lane alpha; exp2(0)==1.0 exactly, so in steady
      //      state (running max unchanged in every lane) skip the 32 pk_muls ----
      if (__any(al != 1.0f)) {
#pragma unroll
        for (int t = 0; t < 8; ++t) {
#pragma unroll
          for (int j = 0; j < 8; ++j) o[t][j] *= al;
        }
      }

      // ---- P^T B-fragments: pack f16 pairs + half-wave exchange (no LDS) ----
      union H2I { v2fp h; int i; };
      union PBV { v16h v; int i[8]; };
      int pkt[4][4];
#pragma unroll
      for (int t = 0; t < 4; ++t) {
#pragma unroll
        for (int m = 0; m < 4; ++m) {
          H2I u;
          u.h = __builtin_amdgcn_cvt_pkrtz(s[t][2 * m], s[t][2 * m + 1]);
          pkt[t][m] = u.i;
        }
      }
      v16h pb[2];
#pragma unroll
      for (int kc = 0; kc < 2; ++kc) {
        PBV w;
#pragma unroll
        for (int m = 0; m < 4; ++m) {
          const int a  = pkt[2 * kc][m];      // kv chunk, low-half rows
          const int b  = pkt[2 * kc + 1][m];  // kv chunk, high-half rows
          const int sa = __shfl_xor(a, 16);
          const int sb = __shfl_xor(b, 16);
          w.i[m]     = hi ? sb : a;   // k = 32kc + 16hi + 0..7
          w.i[4 + m] = hi ? b  : sa;  // k = 32kc + 16hi + 8..15
        }
        pb[kc] = w.v;
      }

      // ---- O^T += V^T(16x32) * P^T(32x16): 16 WMMAs ----
#pragma unroll
      for (int t = 0; t < 8; ++t) {
#pragma unroll
        for (int kc = 0; kc < 2; ++kc) {
          v16h va = ldsA(&sVt[t * 16 + nlo][kc * 32 + khalf]);
          o[t] = wmma16(va, pb[kc], o[t]);
        }
      }
    }
    __syncthreads();
  }

  // ---- epilogue: per-lane 1/l, contiguous float4 stores of this lane's row ----
  const float il = 1.0f / l_w;
  float* orow = O + base + (size_t)r * DH;
#pragma unroll
  for (int t = 0; t < 8; ++t) {
    float4 y0, y1;
    y0.x = o[t][0] * il; y0.y = o[t][1] * il;
    y0.z = o[t][2] * il; y0.w = o[t][3] * il;
    y1.x = o[t][4] * il; y1.y = o[t][5] * il;
    y1.z = o[t][6] * il; y1.w = o[t][7] * il;
    *(float4*)(orow + t * 16 + dsel)     = y0;
    *(float4*)(orow + t * 16 + dsel + 4) = y1;
  }
}

extern "C" void kernel_launch(void* const* d_in, const int* in_sizes, int n_in,
                              void* d_out, int out_size, void* d_ws, size_t ws_size,
                              hipStream_t stream) {
  const float* q = (const float*)d_in[0];
  const float* k = (const float*)d_in[1];
  const float* v = (const float*)d_in[2];
  float* out = (float*)d_out;

  const int bh = in_sizes[0] / (SEQ * DH);   // B*H = 32
  dim3 grid(SEQ / BQ, bh);                   // (16, 32)
  dim3 block(256);                           // 8 waves (wave32)
  fa2_causal_wmma_kernel<<<grid, block, 0, stream>>>(q, k, v, out);
}
